// GATJKClassifier_54537494724782
// MI455X (gfx1250) — matrix operand order repaired and checked by reading, not compiled, not run on hardware
//
#include <hip/hip_runtime.h>
#include <hip/hip_bf16.h>

// ---------------------------------------------------------------------------
// Constants from the reference
// ---------------------------------------------------------------------------
#define NN      50000
#define EE      800000
#define GG      50
#define HID     32
#define HEADS   8
#define HC      256          // HEADS*HID
#define TEXT    768
#define VIS     512
#define AUD     128
#define NV      10000
#define NA      10000
#define NPG     1000         // nodes per graph (N/G)
#define EN      (EE + NN)    // edges incl. self loops
#define FINAL   800          // 3*HC + HID
#define EPS     1e-5f

typedef __bf16 bf16_t;
typedef bf16_t v16bf __attribute__((ext_vector_type(16)));
typedef float  v8f   __attribute__((ext_vector_type(8)));

// round-to-nearest-even fp32 -> bf16, two values packed into one dword
__device__ __forceinline__ unsigned pk2bf(float lo, float hi) {
    unsigned a = __float_as_uint(lo), b = __float_as_uint(hi);
    a += 0x7FFFu + ((a >> 16) & 1u);
    b += 0x7FFFu + ((b >> 16) & 1u);
    return (a >> 16) | (b & 0xFFFF0000u);
}

__device__ __forceinline__ float wave_sum(float v) {
    #pragma unroll
    for (int m = 16; m > 0; m >>= 1) v += __shfl_xor(v, m, 32);
    return v;
}

// ---------------------------------------------------------------------------
// WMMA GEMM:  C[M,Nc] = A[M,K] @ W[K,Nc] + bias   (fp32 in/out, bf16 WMMA core)
// Block: 256 threads = 8 waves, tile 64(M) x 32(N), K-chunks of 32.
// LDS tiles are stored in *fragment order*: [tile][lane] -> v16bf, so each
// lane fetches its whole WMMA operand with contiguous 32-byte LDS loads.
// ---------------------------------------------------------------------------
#define BM 64
#define BN 32
#define KC 32

__global__ __launch_bounds__(256)
void wmma_gemm_bias(const float* __restrict__ A, int lda,
                    const float* __restrict__ W,      // [K, Nc] row-major
                    const float* __restrict__ bias,   // [Nc]
                    float* __restrict__ C, int ldc,
                    int M, int K, int Nc)
{
    __shared__ v16bf aF[4][32];   // A fragments: [row tile][lane]   (4 KB)
    __shared__ v16bf bF[2][32];   // B fragments: [col tile][lane]   (2 KB)

    const int tid  = threadIdx.x;
    const int lane = tid & 31;
    const int wave = tid >> 5;          // 0..7
    const int wm   = wave & 3;          // 4 row tiles of 16
    const int wn   = wave >> 2;         // 2 col tiles of 16
    const int rowBase = blockIdx.x * BM;
    const int colBase = blockIdx.y * BN;

    // --- A staging geometry: thread t loads row r, 8 consecutive K at c0 ---
    const int ar  = tid >> 2;            // 0..63
    const int ac0 = (tid & 3) * 8;       // 0,8,16,24
    const int a_gr = rowBase + ar;
    // destination: all 8 K share (blk, khalf) -> consecutive j in one lane
    const int a_wm    = ar >> 4;
    const int a_lane  = ((ac0 >> 3) & 1) * 16 + (ar & 15);
    const int a_qidx  = ac0 >> 4;        // which uint4 half of the 32B fragment

    // --- B staging geometry: thread t loads column n, 4 consecutive K at kb ---
    const int bn  = tid & 31;
    const int bkb = (tid >> 5) * 4;      // 0..28
    const int b_wn   = bn >> 4;
    const int b_lane = ((bkb >> 3) & 1) * 16 + (bn & 15);
    const int b_qidx = ((bkb >> 4) << 1) + ((bkb & 7) >> 2);  // which uint2 quarter of the fragment

    v8f acc = {};

    for (int k0 = 0; k0 < K; k0 += KC) {
        // ---- stage A: two coalesced float4 loads -> one 16B LDS store ----
        float4 f0 = {0,0,0,0}, f1 = {0,0,0,0};
        if (a_gr < M) {
            const float4* src = reinterpret_cast<const float4*>(A + (long)a_gr*lda + k0 + ac0);
            f0 = src[0]; f1 = src[1];
        }
        uint4 qa;
        qa.x = pk2bf(f0.x, f0.y); qa.y = pk2bf(f0.z, f0.w);
        qa.z = pk2bf(f1.x, f1.y); qa.w = pk2bf(f1.z, f1.w);
        reinterpret_cast<uint4*>(&aF[a_wm][a_lane])[a_qidx] = qa;

        // ---- stage B: 4 coalesced rows of W -> one 8B LDS store ----
        const float* wp = W + (long)(k0 + bkb)*Nc + colBase + bn;
        float w0 = wp[0], w1 = wp[Nc], w2 = wp[2*Nc], w3 = wp[3*Nc];
        uint2 qb; qb.x = pk2bf(w0, w1); qb.y = pk2bf(w2, w3);
        reinterpret_cast<uint2*>(&bF[b_wn][b_lane])[b_qidx] = qb;

        __syncthreads();
        v16bf af = aF[wm][lane];          // 32B contiguous LDS load
        v16bf bv = bF[wn][lane];          // 32B contiguous LDS load
        acc = __builtin_amdgcn_wmma_f32_16x16x32_bf16(
                  false, af, false, bv, (short)0, acc, false, false);
        __syncthreads();
    }

    // C/D layout: lane 0-15 -> M = r, lane 16-31 -> M = r+8 ; N = lane&15
    const int col = colBase + wn*16 + (lane & 15);
    const float bvb = bias ? bias[col] : 0.f;
    #pragma unroll
    for (int r = 0; r < 8; ++r) {
        int row = rowBase + wm*16 + ((lane >> 4) << 3) + r;
        if (row < M) C[(long)row*ldc + col] = acc[r] + bvb;
    }
}

// ---------------------------------------------------------------------------
// Generic float fill
// ---------------------------------------------------------------------------
__global__ void k_fill(float* __restrict__ p, long n, float v) {
    long i = (long)blockIdx.x * blockDim.x + threadIdx.x;
    if (i < n) p[i] = v;
}

// ---------------------------------------------------------------------------
// Modality fixup: final_x[idx[i]] = relu(LN(proj[i])) * rowmean(x_raw[idx[i]])
// One wave (32 lanes) per node; lane == feature.
// ---------------------------------------------------------------------------
__global__ __launch_bounds__(256)
void modal_fixup_k(const float* __restrict__ proj,     // [Nm,32] (bias included)
                   const float* __restrict__ gamma,
                   const float* __restrict__ beta,
                   const float* __restrict__ xraw,     // [N,768]
                   const int*   __restrict__ idx, int Nm,
                   float* __restrict__ final_x)
{
    int node = blockIdx.x * (blockDim.x >> 5) + (threadIdx.x >> 5);
    int f = threadIdx.x & 31;
    if (node >= Nm) return;

    float t  = proj[(long)node*HID + f];
    float mu = wave_sum(t) * (1.f/32.f);
    float d  = t - mu;
    float var = wave_sum(d*d) * (1.f/32.f);
    float h = gamma[f]*d*rsqrtf(var + EPS) + beta[f];
    h = h > 0.f ? h : 0.f;

    int row = idx[node];
    const float* xr = xraw + (long)row*TEXT;
    float s = 0.f;
    #pragma unroll
    for (int k = 0; k < TEXT/32; ++k) s += xr[f + 32*k];
    s = wave_sum(s) * (1.f/(float)TEXT);

    final_x[(long)row*HID + f] = h * s;
}

// ---------------------------------------------------------------------------
// Edge kernels. Edge e<EE: src=ei[e], dst=ei[EE+e]; e>=EE: self loop.
// ---------------------------------------------------------------------------
__device__ __forceinline__ void edge_sd(const int* ei, long e, int& s, int& d) {
    if (e < EE) { s = ei[e]; d = ei[EE + e]; }
    else        { s = d = (int)(e - EE); }
}

__global__ void edge_logits_k(const float* __restrict__ xl, const float* __restrict__ xr,
                              const float* __restrict__ att, const int* __restrict__ ei,
                              int Hh, int F,
                              float* __restrict__ logits, float* __restrict__ mmax)
{
    long gid = (long)blockIdx.x * blockDim.x + threadIdx.x;
    if (gid >= (long)EN * Hh) return;
    int h = (int)(gid % Hh);
    long e = gid / Hh;
    int s, d; edge_sd(ei, e, s, d);

    const float4* pl = reinterpret_cast<const float4*>(xl + (long)s*F + h*HID);
    const float4* pr = reinterpret_cast<const float4*>(xr + (long)d*F + h*HID);
    const float4* pa = reinterpret_cast<const float4*>(att + h*HID);
    float acc = 0.f;
    #pragma unroll
    for (int c = 0; c < HID/4; ++c) {
        float4 a = pl[c], b = pr[c], wv = pa[c];
        float v;
        v = a.x + b.x; v = v > 0.f ? v : 0.2f*v; acc += wv.x * v;
        v = a.y + b.y; v = v > 0.f ? v : 0.2f*v; acc += wv.y * v;
        v = a.z + b.z; v = v > 0.f ? v : 0.2f*v; acc += wv.z * v;
        v = a.w + b.w; v = v > 0.f ? v : 0.2f*v; acc += wv.w * v;
    }
    logits[e*Hh + h] = acc;

    float* addr = mmax + (long)d*Hh + h;   // ordered-int float atomic max
    if (acc >= 0.f) atomicMax((int*)addr, __float_as_int(acc));
    else            atomicMin((unsigned int*)addr, __float_as_uint(acc));
}

__global__ void edge_softmax_k(float* __restrict__ logits,
                               const float* __restrict__ mmax,
                               float* __restrict__ denom,
                               const int* __restrict__ ei, int Hh)
{
    long gid = (long)blockIdx.x * blockDim.x + threadIdx.x;
    if (gid >= (long)EN * Hh) return;
    int h = (int)(gid % Hh);
    long e = gid / Hh;
    int s, d; edge_sd(ei, e, s, d); (void)s;
    float p = __expf(logits[e*Hh + h] - mmax[(long)d*Hh + h]);
    logits[e*Hh + h] = p;
    atomicAdd(&denom[(long)d*Hh + h], p);
}

__global__ void edge_alpha_k(float* __restrict__ logits,
                             const float* __restrict__ denom,
                             const int* __restrict__ ei, int Hh)
{
    long gid = (long)blockIdx.x * blockDim.x + threadIdx.x;
    if (gid >= (long)EN * Hh) return;
    int h = (int)(gid % Hh);
    long e = gid / Hh;
    int s, d; edge_sd(ei, e, s, d); (void)s;
    logits[e*Hh + h] /= denom[(long)d*Hh + h];
}

__global__ void edge_aggregate_k(const float* __restrict__ alpha,
                                 const float* __restrict__ xl,
                                 const int* __restrict__ ei, int Hh, int F,
                                 float* __restrict__ agg)
{
    long gid = (long)blockIdx.x * blockDim.x + threadIdx.x;
    if (gid >= (long)EN * F) return;
    int f = (int)(gid % F);
    long e = gid / F;
    int h = f >> 5;
    int s, d; edge_sd(ei, e, s, d);
    float a = alpha[e*Hh + h];
    atomicAdd(&agg[(long)d*F + f], a * xl[(long)s*F + f]);
}

// ---------------------------------------------------------------------------
// GraphNorm (graphs are contiguous 1000-node blocks)
// pre = agg + conv_bias (+ residual);  stats then apply (+ optional ELU)
// ---------------------------------------------------------------------------
__global__ __launch_bounds__(256)
void graph_stats_k(const float* __restrict__ agg, const float* __restrict__ cbias,
                   const float* __restrict__ res, int ldr, int F, int chunks,
                   float* __restrict__ s1, float* __restrict__ s2)
{
    int g = blockIdx.x, chunk = blockIdx.y, j = threadIdx.x;
    if (j >= F) return;
    int per = NPG / chunks;
    long n0 = (long)g*NPG + (long)chunk*per;
    float a = 0.f, b = 0.f;
    for (int i = 0; i < per; ++i) {
        long n = n0 + i;
        float v = agg[n*F + j] + cbias[j];
        if (res) v += res[n*(long)ldr + j];
        a += v; b += v*v;
    }
    atomicAdd(&s1[g*F + j], a);
    atomicAdd(&s2[g*F + j], b);
}

__global__ void graph_apply_k(const float* __restrict__ agg, const float* __restrict__ cbias,
                              const float* __restrict__ res, int ldr,
                              const float* __restrict__ s1, const float* __restrict__ s2,
                              const float* __restrict__ gn_g, const float* __restrict__ gn_b,
                              const float* __restrict__ gn_ms,
                              int F, int doElu,
                              float* __restrict__ out, int ldo)
{
    long gid = (long)blockIdx.x * blockDim.x + threadIdx.x;
    if (gid >= (long)NN * F) return;
    long n = gid / F;
    int  j = (int)(gid % F);
    int  g = (int)(n / NPG);

    float v = agg[n*F + j] + cbias[j];
    if (res) v += res[n*(long)ldr + j];

    const float inv = 1.f/(float)NPG;
    float mean = s1[g*F + j] * inv;
    float ex2  = s2[g*F + j] * inv;
    float ms   = gn_ms[j];
    float centered = v - mean*ms;
    float var = ex2 - mean*mean*ms*(2.f - ms);   // E[(x - ms*mean)^2]
    float y = gn_g[j]*centered*rsqrtf(var + EPS) + gn_b[j];
    if (doElu) y = y > 0.f ? y : (__expf(y) - 1.f);
    out[n*(long)ldo + j] = y;
}

// ---------------------------------------------------------------------------
// Final head: out[g] = xcat[ptr[g]] . lin_W + lin_b
// ---------------------------------------------------------------------------
__global__ __launch_bounds__(256)
void final_lin_k(const float* __restrict__ xcat, const int* __restrict__ ptr,
                 const float* __restrict__ W, const float* __restrict__ b,
                 float* __restrict__ out)
{
    __shared__ float red[256];
    int g = blockIdx.x, t = threadIdx.x;
    long node = ptr[g];
    float s = 0.f;
    for (int f = t; f < FINAL; f += 256) s += xcat[node*FINAL + f] * W[f];
    red[t] = s; __syncthreads();
    for (int w = 128; w > 0; w >>= 1) { if (t < w) red[t] += red[t + w]; __syncthreads(); }
    if (t == 0) out[g] = red[0] + b[0];
}

// ---------------------------------------------------------------------------
// Host driver
// ---------------------------------------------------------------------------
static inline dim3 g1(long n, int bs) { return dim3((unsigned)((n + bs - 1) / bs)); }

extern "C" void kernel_launch(void* const* d_in, const int* in_sizes, int n_in,
                              void* d_out, int out_size, void* d_ws, size_t ws_size,
                              hipStream_t stream)
{
    const float* x        = (const float*)d_in[0];
    const float* x_vision = (const float*)d_in[1];
    const float* x_audio  = (const float*)d_in[2];
    const int*   ei       = (const int*)  d_in[3];
    const int*   ptrs     = (const int*)  d_in[5];
    const int*   vidx     = (const int*)  d_in[6];
    const int*   aidx     = (const int*)  d_in[7];
    const float* text_W   = (const float*)d_in[8];
    const float* text_b   = (const float*)d_in[9];
    const float* vis_W    = (const float*)d_in[10];
    const float* vis_b    = (const float*)d_in[11];
    const float* vis_g    = (const float*)d_in[12];
    const float* vis_be   = (const float*)d_in[13];
    const float* aud_W    = (const float*)d_in[14];
    const float* aud_b    = (const float*)d_in[15];
    const float* aud_g    = (const float*)d_in[16];
    const float* aud_be   = (const float*)d_in[17];
    const float* lin_W    = (const float*)d_in[54];
    const float* lin_b    = (const float*)d_in[55];

    // Workspace layout (floats)
    float* w      = (float*)d_ws;
    float* finalx = w;                        // N*32
    float* xcat   = finalx + (long)NN*HID;    // N*800
    float* xl     = xcat   + (long)NN*FINAL;  // N*256
    float* xr     = xl     + (long)NN*HC;     // N*256
    float* agg    = xr     + (long)NN*HC;     // N*256
    float* logits = agg    + (long)NN*HC;     // EN*8
    float* mmax   = logits + (long)EN*HEADS;  // N*8
    float* denom  = mmax   + (long)NN*HEADS;  // N*8
    float* s1     = denom  + (long)NN*HEADS;  // G*256
    float* s2     = s1     + (long)GG*HC;     // G*256
    float* projt  = agg;                      // reuse for modality proj [10000,32]

    // ---- text projection: finalx = x @ text_W + text_b (WMMA) ----
    wmma_gemm_bias<<<dim3((NN+BM-1)/BM, HID/BN), 256, 0, stream>>>(
        x, TEXT, text_W, text_b, finalx, HID, NN, TEXT, HID);

    // ---- vision / audio overwrite ----
    wmma_gemm_bias<<<dim3((NV+BM-1)/BM, HID/BN), 256, 0, stream>>>(
        x_vision, VIS, vis_W, vis_b, projt, HID, NV, VIS, HID);
    modal_fixup_k<<<g1((long)NV*32, 256), 256, 0, stream>>>(
        projt, vis_g, vis_be, x, vidx, NV, finalx);

    wmma_gemm_bias<<<dim3((NA+BM-1)/BM, HID/BN), 256, 0, stream>>>(
        x_audio, AUD, aud_W, aud_b, projt, HID, NA, AUD, HID);
    modal_fixup_k<<<g1((long)NA*32, 256), 256, 0, stream>>>(
        projt, aud_g, aud_be, x, aidx, NA, finalx);

    // ---- one GATv2 + GraphNorm (+ residual, + optional ELU) layer ----
    auto gat_layer = [&](const float* in, int lda, int inDim, int convBase,
                         int Hh, const float* res, int ldr, int normBase,
                         int doElu, float* outCol) {
        const float* Wl  = (const float*)d_in[convBase + 0];
        const float* bl  = (const float*)d_in[convBase + 1];
        const float* Wr  = (const float*)d_in[convBase + 2];
        const float* br  = (const float*)d_in[convBase + 3];
        const float* att = (const float*)d_in[convBase + 4];
        const float* cb  = (const float*)d_in[convBase + 5];
        const float* ng  = (const float*)d_in[normBase + 0];
        const float* nb  = (const float*)d_in[normBase + 1];
        const float* nms = (const float*)d_in[normBase + 2];
        const int F = Hh * HID;

        dim3 gemm_grid((NN+BM-1)/BM, F/BN);
        wmma_gemm_bias<<<gemm_grid, 256, 0, stream>>>(in, lda, Wl, bl, xl, F, NN, inDim, F);
        wmma_gemm_bias<<<gemm_grid, 256, 0, stream>>>(in, lda, Wr, br, xr, F, NN, inDim, F);

        k_fill<<<g1((long)NN*Hh, 256), 256, 0, stream>>>(mmax, (long)NN*Hh, -__builtin_inff());
        k_fill<<<g1((long)NN*Hh, 256), 256, 0, stream>>>(denom, (long)NN*Hh, 0.f);
        k_fill<<<g1((long)NN*F, 256), 256, 0, stream>>>(agg, (long)NN*F, 0.f);

        edge_logits_k   <<<g1((long)EN*Hh, 256), 256, 0, stream>>>(xl, xr, att, ei, Hh, F, logits, mmax);
        edge_softmax_k  <<<g1((long)EN*Hh, 256), 256, 0, stream>>>(logits, mmax, denom, ei, Hh);
        edge_alpha_k    <<<g1((long)EN*Hh, 256), 256, 0, stream>>>(logits, denom, ei, Hh);
        edge_aggregate_k<<<g1((long)EN*F,  256), 256, 0, stream>>>(logits, xl, ei, Hh, F, agg);

        k_fill<<<g1((long)GG*F, 256), 256, 0, stream>>>(s1, (long)GG*F, 0.f);
        k_fill<<<g1((long)GG*F, 256), 256, 0, stream>>>(s2, (long)GG*F, 0.f);
        graph_stats_k<<<dim3(GG, 5), 256, 0, stream>>>(agg, cb, res, ldr, F, 5, s1, s2);
        graph_apply_k<<<g1((long)NN*F, 256), 256, 0, stream>>>(
            agg, cb, res, ldr, s1, s2, ng, nb, nms, F, doElu, outCol, FINAL);
    };

    // L1: in=finalx[32] -> xcat[:,0:256), ELU, no residual
    gat_layer(finalx, HID, HID, 18, HEADS, nullptr, 0, 42, 1, xcat + 0);
    // L2: in=h1 -> xcat[:,256:512), residual h1, ELU
    gat_layer(xcat + 0,   FINAL, HC, 24, HEADS, xcat + 0,   FINAL, 45, 1, xcat + HC);
    // L3: in=h2 -> xcat[:,512:768), residual h2, ELU
    gat_layer(xcat + HC,  FINAL, HC, 30, HEADS, xcat + HC,  FINAL, 48, 1, xcat + 2*HC);
    // L4: in=h3 -> xcat[:,768:800), heads=1, no residual, no ELU
    gat_layer(xcat + 2*HC, FINAL, HC, 36, 1, nullptr, 0, 51, 0, xcat + 3*HC);

    // ---- classifier head ----
    final_lin_k<<<GG, 256, 0, stream>>>(xcat, ptrs, lin_W, lin_b, (float*)d_out);

    (void)in_sizes; (void)n_in; (void)out_size; (void)ws_size;
}
